// Mamba4CastForecaster_47339129536577
// MI455X (gfx1250) — matrix operand order repaired and checked by reading, not compiled, not run on hardware
//
#include <hip/hip_runtime.h>
#include <hip/hip_bf16.h>
#include <math.h>

// ---------------------------------------------------------------------------
// Mamba4Cast forecaster for MI455X (gfx1250, wave32).
//  - Big GEMMs: v_wmma_f32_16x16x32_bf16, 32x64 wave tile (8 WMMA / k-step,
//    1.5 b128 loads per WMMA).
//  - Selective scan: register-state VALU; per-chunk B/C coefficients staged
//    into LDS by the Tensor Data Mover (tensor_load_to_lds + s_wait_tensorcnt).
// ---------------------------------------------------------------------------

typedef __attribute__((ext_vector_type(16))) __bf16 v16bf;
typedef __attribute__((ext_vector_type(8)))  __bf16 v8bf;
typedef __attribute__((ext_vector_type(8)))  float  v8f;
typedef __attribute__((ext_vector_type(4)))  unsigned int v4u;
typedef __attribute__((ext_vector_type(8)))  int v8i;
typedef __attribute__((ext_vector_type(4)))  int v4i;

constexpr int Bc  = 8;
constexpr int Lc  = 1024;
constexpr int DMc = 256;
constexpr int DIc = 512;     // EXP*DM
constexpr int DSc = 16;
constexpr int DCc = 4;
constexpr int DTRc= 16;
constexpr int NLc = 4;
constexpr int XDc = DTRc + 2*DSc;   // 48
constexpr int Mrows = Bc * Lc;      // 8192

#define DEV static __device__ __forceinline__

DEV float siluf(float x)     { return x / (1.f + __expf(-x)); }
DEV float softplusf(float x) { return (x > 20.f) ? x : log1pf(__expf(x)); }
DEV float geluf(float x)     { return 0.5f * x * (1.f + erff(x * 0.70710678118654752f)); }

// ---------------------------------------------------------------------------
// TDM: 2D strided tile (global -> LDS), data_size = 4 bytes.
// D# packing per CDNA5 ISA (groups 0/1; 2D => groups
// 2/3 zero). This toolchain exposes the 6-arg builtin form.
// ---------------------------------------------------------------------------
DEV void tdm_load_2d_f32(unsigned lds_off, const void* gaddr,
                         unsigned tensor_d0, unsigned tensor_d1,
                         unsigned tile_d0,  unsigned tile_d1,
                         unsigned stride0_elems) {
  unsigned long long ga = (unsigned long long)(uintptr_t)gaddr;
  v4u g0;
  g0[0] = 1u;                                              // count=1 (user D#)
  g0[1] = lds_off;                                         // lds_addr (bytes)
  g0[2] = (unsigned)(ga & 0xFFFFFFFFu);                    // global_addr[31:0]
  g0[3] = (unsigned)((ga >> 32) & 0x01FFFFFFu) | (2u << 30); // [56:32] | type=2
  v8i g1;
  g1[0] = (int)(2u << 16);                                 // data_size=2 (4B)
  g1[1] = (int)((tensor_d0 & 0xFFFFu) << 16);              // tensor_dim0[15:0]
  g1[2] = (int)(((tensor_d0 >> 16) & 0xFFFFu) |            // tensor_dim0[31:16]
                ((tensor_d1 & 0xFFFFu) << 16));            // tensor_dim1[15:0]
  g1[3] = (int)(((tensor_d1 >> 16) & 0xFFFFu) |            // tensor_dim1[31:16]
                ((tile_d0 & 0xFFFFu) << 16));              // tile_dim0
  g1[4] = (int)(tile_d1 & 0xFFFFu);                        // tile_dim1, tile_dim2=0
  g1[5] = (int)stride0_elems;                              // dim0_stride[31:0]
  g1[6] = 0;                                               // dim0_stride[47:32]
  g1[7] = 0;
  v4i z4 = {0, 0, 0, 0};
  v8i z8 = {0, 0, 0, 0, 0, 0, 0, 0};
  __builtin_amdgcn_tensor_load_to_lds(g0, g1, z4, z4, z8, 0);
}

// ---------------------------------------------------------------------------
// f32 -> bf16 converter (weights / activations)
// ---------------------------------------------------------------------------
__global__ void f32_to_bf16_k(const float* __restrict__ in, __bf16* __restrict__ out, int n) {
  int i = blockIdx.x * blockDim.x + threadIdx.x;
  int stride = gridDim.x * blockDim.x;
  for (; i < n; i += stride) out[i] = (__bf16)in[i];
}

// ---------------------------------------------------------------------------
// Per-batch mean / std (ddof=1) over L for the NF=1 input series
// ---------------------------------------------------------------------------
__global__ void batch_stats_k(const float* __restrict__ x, float* __restrict__ stats) {
  __shared__ float r1[256], r2[256];
  int b = blockIdx.x, tid = threadIdx.x;
  float s1 = 0.f, s2 = 0.f;
  for (int t = tid; t < Lc; t += 256) {
    float v = x[(size_t)b * Lc + t];
    s1 += v; s2 += v * v;
  }
  r1[tid] = s1; r2[tid] = s2; __syncthreads();
  for (int s = 128; s > 0; s >>= 1) {
    if (tid < s) { r1[tid] += r1[tid + s]; r2[tid] += r2[tid + s]; }
    __syncthreads();
  }
  if (tid == 0) {
    float mean = r1[0] / (float)Lc;
    float var  = (r2[0] - (float)Lc * mean * mean) / (float)(Lc - 1);
    var = fmaxf(var, 0.f);
    float sd = fmaxf(sqrtf(var), 1e-6f);
    stats[b * 2 + 0] = mean;
    stats[b * 2 + 1] = sd;
  }
}

// ---------------------------------------------------------------------------
// h[b,t,:] = xnorm * inp_w[:,0] + inp_b + PE(t,:)   (one block per token row)
// ---------------------------------------------------------------------------
__global__ void init_h_k(const float* __restrict__ x, const float* __restrict__ stats,
                         const float* __restrict__ inp_w, const float* __restrict__ inp_b,
                         float* __restrict__ h) {
  int row = blockIdx.x;            // b*L + t
  int c   = threadIdx.x;           // 0..255
  int b = row / Lc, t = row % Lc;
  float xn = (x[row] - stats[b * 2]) / stats[b * 2 + 1];
  int i = c >> 1;
  float div = __expf((float)(2 * i) * (-9.210340371976184f / (float)DMc)); // -ln(1e4)/DM
  float ang = (float)t * div;
  float pe = (c & 1) ? cosf(ang) : sinf(ang);
  h[(size_t)row * DMc + c] = xn * inp_w[c] + inp_b[c] + pe;
}

// ---------------------------------------------------------------------------
// Row LayerNorm(DM=256) -> bf16 (feeds WMMA GEMM)
// ---------------------------------------------------------------------------
__global__ void ln_rows_k(const float* __restrict__ h, const float* __restrict__ g,
                          const float* __restrict__ bb, __bf16* __restrict__ out) {
  __shared__ float red[256];
  int row = blockIdx.x, tid = threadIdx.x;
  float v = h[(size_t)row * DMc + tid];
  red[tid] = v; __syncthreads();
  for (int s = 128; s > 0; s >>= 1) { if (tid < s) red[tid] += red[tid + s]; __syncthreads(); }
  float mean = red[0] / (float)DMc; __syncthreads();
  float d = v - mean;
  red[tid] = d * d; __syncthreads();
  for (int s = 128; s > 0; s >>= 1) { if (tid < s) red[tid] += red[tid + s]; __syncthreads(); }
  float var = red[0] / (float)DMc;
  float o = d * rsqrtf(var + 1e-5f) * g[tid] + bb[tid];
  out[(size_t)row * DMc + tid] = (__bf16)o;
}

// ---------------------------------------------------------------------------
// WMMA GEMM:  C[M,N](f32) (+)= A[M,K](bf16) @ B[N,K](bf16)^T
// 32(M) x 64(N) tile per wave: 2 A frags x 4 B frags -> 8 WMMA per k-step.
// A/B k order contiguous (lane half h holds k = 16h..16h+15); f32 C/D per CDNA5 ISA 7.12.2.
// ---------------------------------------------------------------------------
template <bool ACC>
__global__ __launch_bounds__(128)
void gemm_bf16_wmma_k(const __bf16* __restrict__ A, const __bf16* __restrict__ Bw,
                      float* __restrict__ C, int M, int N, int K) {
  const int lane = threadIdx.x & 31;
  const int wave = threadIdx.x >> 5;
  const int tile = blockIdx.x * 4 + wave;
  const int tilesN = N >> 6;                 // 64-wide N tiles
  const int tm = tile / tilesN;
  const int tn = tile - tm * tilesN;
  const int m0 = tm << 5, n0 = tn << 6;

  const int r15 = lane & 15;
  const int hi  = (lane >> 4) & 1;
  const int koffA = hi * 16;                 // A: lane group picks K+16..31
  const int koffB = hi * 16;                 // B: lane group picks K+16..31
  const int cm = hi * 8;                     // C/D: M = r + 8*hi

  const __bf16* aP[2];
  aP[0] = A + (size_t)(m0 + r15) * (size_t)K + koffA;
  aP[1] = A + (size_t)(m0 + 16 + r15) * (size_t)K + koffA;
  const __bf16* bP[4];
#pragma unroll
  for (int j = 0; j < 4; ++j)
    bP[j] = Bw + (size_t)(n0 + j * 16 + r15) * (size_t)K + koffB;

  v8f c[2][4];
#pragma unroll
  for (int i = 0; i < 2; ++i)
#pragma unroll
    for (int j = 0; j < 4; ++j) {
      if (ACC) {
#pragma unroll
        for (int r = 0; r < 8; ++r)
          c[i][j][r] = C[(size_t)(m0 + i * 16 + cm + r) * (size_t)N + (n0 + j * 16 + r15)];
      } else {
#pragma unroll
        for (int r = 0; r < 8; ++r) c[i][j][r] = 0.f;
      }
    }

  for (int k0 = 0; k0 < K; k0 += 32) {
    v16bf a[2], b[4];
#pragma unroll
    for (int i = 0; i < 2; ++i) {
      v8bf lo = *(const v8bf*)(aP[i] + k0);        // K = k0+koffA .. +7
      v8bf hi8 = *(const v8bf*)(aP[i] + k0 + 8);   // K = k0+koffA+8 .. +15
#pragma unroll
      for (int e = 0; e < 8; ++e) { a[i][e] = lo[e]; a[i][e + 8] = hi8[e]; }
    }
#pragma unroll
    for (int j = 0; j < 4; ++j)
      b[j] = *(const v16bf*)(bP[j] + k0);
#pragma unroll
    for (int i = 0; i < 2; ++i)
#pragma unroll
      for (int j = 0; j < 4; ++j)
        c[i][j] = __builtin_amdgcn_wmma_f32_16x16x32_bf16(
            false, a[i], false, b[j], (short)0, c[i][j], false, false);
  }

#pragma unroll
  for (int i = 0; i < 2; ++i)
#pragma unroll
    for (int j = 0; j < 4; ++j)
#pragma unroll
      for (int r = 0; r < 8; ++r)
        C[(size_t)(m0 + i * 16 + cm + r) * (size_t)N + (n0 + j * 16 + r15)] = c[i][j][r];
}

// ---------------------------------------------------------------------------
// Causal depthwise conv (DC=4) on first DI cols of xz, + bias, SiLU -> xc
// ---------------------------------------------------------------------------
__global__ void conv_silu_k(const float* __restrict__ xz, const float* __restrict__ cw,
                            const float* __restrict__ cb, float* __restrict__ xc) {
  int idx = blockIdx.x * blockDim.x + threadIdx.x;     // over Mrows*DI
  if (idx >= Mrows * DIc) return;
  int d = idx & (DIc - 1);
  int m = idx >> 9;                                    // b*L + t
  int t = m & (Lc - 1);
  float acc = cb[d];
#pragma unroll
  for (int j = 0; j < DCc; ++j) {
    int ts = t - (DCc - 1) + j;
    if (ts >= 0)
      acc += cw[d * DCc + j] * xz[((size_t)(m - (DCc - 1) + j)) * (size_t)(2 * DIc) + d];
  }
  xc[(size_t)m * DIc + d] = siluf(acc);
}

// ---------------------------------------------------------------------------
// xd[m, 0:48] = xc[m,:] @ xproj_w.T   (one block per row; row cached in LDS)
// ---------------------------------------------------------------------------
__global__ void xproj_k(const float* __restrict__ xc, const float* __restrict__ w,
                        float* __restrict__ xd) {
  __shared__ float xr[DIc];
  size_t m = blockIdx.x;
  for (int i = threadIdx.x; i < DIc; i += 64) xr[i] = xc[m * DIc + i];
  __syncthreads();
  if (threadIdx.x < XDc) {
    const float* wr = w + (size_t)threadIdx.x * DIc;
    float acc = 0.f;
    for (int k = 0; k < DIc; ++k) acc += xr[k] * wr[k];
    xd[m * XDc + threadIdx.x] = acc;
  }
}

// ---------------------------------------------------------------------------
// dt[m,d] = softplus( dtr[m,:] @ dtproj_w[d,:] + dtproj_b[d] )
// ---------------------------------------------------------------------------
__global__ void dt_k(const float* __restrict__ xd, const float* __restrict__ w,
                     const float* __restrict__ bias, float* __restrict__ dt) {
  __shared__ float dtr[DTRc];
  size_t m = blockIdx.x;
  int d = threadIdx.x;
  if (d < DTRc) dtr[d] = xd[m * XDc + d];
  __syncthreads();
  float acc = bias[d];
#pragma unroll
  for (int r = 0; r < DTRc; ++r) acc += dtr[r] * w[(size_t)d * DTRc + r];
  dt[m * DIc + d] = softplusf(acc);
}

// ---------------------------------------------------------------------------
// Selective scan: one thread per (b,d); 16-wide state in registers.
// Bm/Cm chunk (64 timesteps x 32 floats, row stride 48) staged into LDS by
// the Tensor Data Mover; dt/xc streams prefetched one chunk ahead.
// ---------------------------------------------------------------------------
__global__ __launch_bounds__(128)
void ssm_scan_k(const float* __restrict__ dt, const float* __restrict__ xc,
                const float* __restrict__ xd, const float* __restrict__ A_log,
                float* __restrict__ y) {
  constexpr int CH = 64;
  __shared__ float bc[CH][2 * DSc];       // [tt][0:16]=B, [16:32]=C
  int d = blockIdx.x * 128 + threadIdx.x;
  int b = blockIdx.y;

  float Arow[DSc];
#pragma unroll
  for (int n = 0; n < DSc; ++n) Arow[n] = -__expf(A_log[(size_t)d * DSc + n]);
  float hs[DSc];
#pragma unroll
  for (int n = 0; n < DSc; ++n) hs[n] = 0.f;

  const unsigned lds_off = (unsigned)(uintptr_t)(void*)&bc[0][0];

  for (int t0 = 0; t0 < Lc; t0 += CH) {
    __syncthreads();
    if (threadIdx.x < 32) {               // wave 0 issues the DMA once
      tdm_load_2d_f32(lds_off,
                      &xd[((size_t)b * Lc + t0) * XDc + DTRc],
                      /*tensor_d0=*/2 * DSc, /*tensor_d1=*/CH,
                      /*tile_d0=*/2 * DSc,   /*tile_d1=*/CH,
                      /*stride0=*/XDc);
      __builtin_amdgcn_s_wait_tensorcnt(0);
    }
    __syncthreads();

    // prefetch next chunk's per-channel streams
    if (t0 + CH < Lc) {
      size_t pnext = ((size_t)b * Lc + (t0 + CH)) * DIc + d;
      __builtin_prefetch(&dt[pnext], 0, 0);
      __builtin_prefetch(&xc[pnext], 0, 0);
    }

    for (int tt = 0; tt < CH; ++tt) {
      size_t off = ((size_t)b * Lc + (t0 + tt)) * DIc + d;
      float dtv = dt[off];
      float xcv = xc[off];
      float acc = 0.f;
#pragma unroll
      for (int n = 0; n < DSc; ++n) {
        hs[n] = __expf(dtv * Arow[n]) * hs[n] + dtv * bc[tt][n] * xcv;
        acc += hs[n] * bc[tt][DSc + n];
      }
      y[off] = acc;
    }
  }
}

// ---------------------------------------------------------------------------
// y16 = bf16( (y + xc*D_p) * silu(z) )      z = xz[..., DI:]
// ---------------------------------------------------------------------------
__global__ void ypost_k(const float* __restrict__ y, const float* __restrict__ xc,
                        const float* __restrict__ xz, const float* __restrict__ Dp,
                        __bf16* __restrict__ y16) {
  int idx = blockIdx.x * blockDim.x + threadIdx.x;
  if (idx >= Mrows * DIc) return;
  int d = idx & (DIc - 1);
  int m = idx >> 9;
  float z = xz[(size_t)m * (2 * DIc) + DIc + d];
  float v = (y[idx] + xc[idx] * Dp[d]) * siluf(z);
  y16[idx] = (__bf16)v;
}

// ---------------------------------------------------------------------------
// Head: LN(last token) -> GELU MLP(256->512->96) -> rescale by (std, mean)
// ---------------------------------------------------------------------------
__global__ void head_k(const float* __restrict__ h, const float* __restrict__ ng,
                       const float* __restrict__ nb, const float* __restrict__ h1w,
                       const float* __restrict__ h1b, const float* __restrict__ h2w,
                       const float* __restrict__ h2b, const float* __restrict__ stats,
                       float* __restrict__ out, int horizon) {
  __shared__ float red[256];
  __shared__ float lastn[DMc];
  __shared__ float f1[2 * DMc];
  int b = blockIdx.x, tid = threadIdx.x;
  const float* last = h + ((size_t)b * Lc + (Lc - 1)) * DMc;
  float v = last[tid];
  red[tid] = v; __syncthreads();
  for (int s = 128; s > 0; s >>= 1) { if (tid < s) red[tid] += red[tid + s]; __syncthreads(); }
  float mean = red[0] / (float)DMc; __syncthreads();
  float d = v - mean;
  red[tid] = d * d; __syncthreads();
  for (int s = 128; s > 0; s >>= 1) { if (tid < s) red[tid] += red[tid + s]; __syncthreads(); }
  float var = red[0] / (float)DMc;
  lastn[tid] = d * rsqrtf(var + 1e-5f) * ng[tid] + nb[tid];
  __syncthreads();
  for (int o = tid; o < 2 * DMc; o += 256) {
    const float* wr = h1w + (size_t)o * DMc;
    float acc = h1b[o];
    for (int k = 0; k < DMc; ++k) acc += lastn[k] * wr[k];
    f1[o] = geluf(acc);
  }
  __syncthreads();
  if (tid < horizon) {
    const float* wr = h2w + (size_t)tid * (2 * DMc);
    float acc = h2b[tid];
    for (int k = 0; k < 2 * DMc; ++k) acc += f1[k] * wr[k];
    out[(size_t)b * horizon + tid] = acc * stats[b * 2 + 1] + stats[b * 2];
  }
}

// ---------------------------------------------------------------------------
// Host driver
// ---------------------------------------------------------------------------
extern "C" void kernel_launch(void* const* d_in, const int* in_sizes, int n_in,
                              void* d_out, int out_size, void* d_ws, size_t ws_size,
                              hipStream_t stream) {
  const float* x       = (const float*)d_in[0];
  const float* ln_g    = (const float*)d_in[1];
  const float* ln_b    = (const float*)d_in[2];
  const float* in_w    = (const float*)d_in[3];
  const float* conv_w  = (const float*)d_in[4];
  const float* conv_b  = (const float*)d_in[5];
  const float* xproj_w = (const float*)d_in[6];
  const float* dtproj_w= (const float*)d_in[7];
  const float* dtproj_b= (const float*)d_in[8];
  const float* A_log   = (const float*)d_in[9];
  const float* D_p     = (const float*)d_in[10];
  const float* out_w   = (const float*)d_in[11];
  const float* inp_w   = (const float*)d_in[12];
  const float* inp_b   = (const float*)d_in[13];
  const float* norm_g  = (const float*)d_in[14];
  const float* norm_b  = (const float*)d_in[15];
  const float* h1_w    = (const float*)d_in[16];
  const float* h1_b    = (const float*)d_in[17];
  const float* h2_w    = (const float*)d_in[18];
  const float* h2_b    = (const float*)d_in[19];
  (void)in_sizes; (void)n_in; (void)ws_size;
  const int horizon = out_size / Bc;   // NF == 1

  char* ws = (char*)d_ws;
  auto carve = [&](size_t bytes) -> char* {
    char* p = ws; ws += (bytes + 255) & ~(size_t)255; return p;
  };
  float*  stats  = (float*) carve(Bc * 2 * sizeof(float));
  float*  h      = (float*) carve((size_t)Mrows * DMc * 4);
  __bf16* u16    = (__bf16*)carve((size_t)Mrows * DMc * 2);
  float*  xz     = (float*) carve((size_t)Mrows * 2 * DIc * 4);
  float*  xc     = (float*) carve((size_t)Mrows * DIc * 4);
  float*  xd     = (float*) carve((size_t)Mrows * XDc * 4);
  float*  dt     = (float*) carve((size_t)Mrows * DIc * 4);
  float*  y      = (float*) carve((size_t)Mrows * DIc * 4);
  __bf16* y16    = (__bf16*)carve((size_t)Mrows * DIc * 2);
  __bf16* inw16  = (__bf16*)carve((size_t)NLc * 2 * DIc * DMc * 2);
  __bf16* outw16 = (__bf16*)carve((size_t)NLc * DMc * DIc * 2);

  f32_to_bf16_k<<<1024, 256, 0, stream>>>(in_w,  inw16,  NLc * 2 * DIc * DMc);
  f32_to_bf16_k<<<512,  256, 0, stream>>>(out_w, outw16, NLc * DMc * DIc);
  batch_stats_k<<<Bc, 256, 0, stream>>>(x, stats);
  init_h_k<<<Mrows, 256, 0, stream>>>(x, stats, inp_w, inp_b, h);

  for (int l = 0; l < NLc; ++l) {
    ln_rows_k<<<Mrows, 256, 0, stream>>>(h, ln_g + l * DMc, ln_b + l * DMc, u16);

    // xz = u @ in_w.T : M=8192, N=1024, K=256 -> (8192/32)*(1024/64)=4096 tiles
    gemm_bf16_wmma_k<false><<<(Mrows / 32) * (2 * DIc / 64) / 4, 128, 0, stream>>>(
        u16, inw16 + (size_t)l * 2 * DIc * DMc, xz, Mrows, 2 * DIc, DMc);

    conv_silu_k<<<(Mrows * DIc) / 256, 256, 0, stream>>>(
        xz, conv_w + (size_t)l * DIc * DCc, conv_b + (size_t)l * DIc, xc);

    xproj_k<<<Mrows, 64, 0, stream>>>(xc, xproj_w + (size_t)l * XDc * DIc, xd);

    dt_k<<<Mrows, DIc, 0, stream>>>(
        xd, dtproj_w + (size_t)l * DIc * DTRc, dtproj_b + (size_t)l * DIc, dt);

    ssm_scan_k<<<dim3(DIc / 128, Bc), 128, 0, stream>>>(
        dt, xc, xd, A_log + (size_t)l * DIc * DSc, y);

    ypost_k<<<(Mrows * DIc) / 256, 256, 0, stream>>>(y, xc, xz, D_p + (size_t)l * DIc, y16);

    // h += y @ out_w.T : M=8192, N=256, K=512 -> (8192/32)*(256/64)=1024 tiles
    gemm_bf16_wmma_k<true><<<(Mrows / 32) * (DMc / 64) / 4, 128, 0, stream>>>(
        y16, outw16 + (size_t)l * DMc * DIc, h, Mrows, DMc, DIc);
  }

  head_k<<<Bc, 256, 0, stream>>>(h, norm_g, norm_b, h1_w, h1_b, h2_w, h2_b,
                                 stats, (float*)d_out, horizon);
}
